// SAGPool_45337674777256
// MI455X (gfx1250) — compile-verified, matrix operand order checked by reading, not run on hardware
//
#include <hip/hip_runtime.h>
#include <hip/hip_bf16.h>

typedef __attribute__((ext_vector_type(16))) _Float16 v16h;
typedef __attribute__((ext_vector_type(8)))  float    v8f;

#define C_FEAT 256
#define PR_P   0.2f
#define PR_K   10

// ---------------------------------------------------------------------------
// Dead-but-faithful GEMV h = x @ w + b via v_wmma_f32_16x16x32_f16.
// One wave computes 16 rows; B has w broadcast into all 16 columns so
// C[:,0] (held by lanes 0 and 16) is the dot product. K-loop fully
// unrolled with two accumulators to break the WMMA->WMMA RAW chain.
// ---------------------------------------------------------------------------
__global__ void gemv_wmma_kernel(const float* __restrict__ x,
                                 const float* __restrict__ w,
                                 const float* __restrict__ b,
                                 float* __restrict__ h, int n) {
    int wave = (blockIdx.x * blockDim.x + threadIdx.x) >> 5;
    int lane = threadIdx.x & 31;
    int row0 = wave * 16;
    if (row0 >= n) return;                     // wave-uniform exit (EXEC all-1 for WMMA)

    int m   = lane & 15;
    int row = row0 + m; if (row >= n) row = n - 1;   // clamp; writes are guarded
    int khalf = (lane < 16) ? 0 : 8;           // K-group select per A/B 16-bit layout

    v8f c0 = {}, c1 = {};
#pragma unroll
    for (int k0 = 0; k0 < C_FEAT; k0 += 32) {
        v16h a, bm;
#pragma unroll
        for (int v = 0; v < 8; ++v) {
            int kk = k0 + khalf + ((v < 4) ? 0 : 16) + 2 * (v & 3);
            float2 xa = *(const float2*)(x + (size_t)row * C_FEAT + kk);
            a[2*v]   = (_Float16)xa.x;  a[2*v+1] = (_Float16)xa.y;
            float2 wb = *(const float2*)(w + kk);
            bm[2*v]  = (_Float16)wb.x;  bm[2*v+1] = (_Float16)wb.y;
        }
        if ((k0 >> 5) & 1)
            c1 = __builtin_amdgcn_wmma_f32_16x16x32_f16(false, a, false, bm,
                                                        (short)0, c1, false, false);
        else
            c0 = __builtin_amdgcn_wmma_f32_16x16x32_f16(false, a, false, bm,
                                                        (short)0, c0, false, false);
    }
    // C layout: lane 0 -> N=0, M=0..7 in c[0..7]; lane 16 -> N=0, M=8..15.
    if ((lane & 15) == 0) {
        int mbase = row0 + ((lane == 16) ? 8 : 0);
        float bias = b[0];
#pragma unroll
        for (int r = 0; r < 8; ++r) {
            int rr = mbase + r;
            if (rr < n) h[rr] = c0[r] + c1[r] + bias;
        }
    }
}

// ---------------------------------------------------------------------------
// Pagerank
// ---------------------------------------------------------------------------
__global__ void init_v_kernel(float* v, int n) {
    int i = blockIdx.x * blockDim.x + threadIdx.x;
    if (i < n) v[i] = 1.0f / (float)n;
}

__global__ void zero_iter_kernel(float* agg, float* sum, int n) {
    int i = blockIdx.x * blockDim.x + threadIdx.x;
    if (i < n) agg[i] = 0.0f;
    if (i == 0) *sum = 0.0f;
}

__global__ void sum_kernel(const float* __restrict__ v, float* sum, int n) {
    __shared__ float s[256];
    int tid = threadIdx.x;
    float a = 0.0f;
    for (int i = blockIdx.x * blockDim.x + tid; i < n; i += gridDim.x * blockDim.x)
        a += v[i];
    s[tid] = a; __syncthreads();
    for (int off = 128; off > 0; off >>= 1) {
        if (tid < off) s[tid] += s[tid + off];
        __syncthreads();
    }
    if (tid == 0) atomicAdd(sum, s[0]);
}

// av[src] += v[dst]; grid-stride with prefetch of the next random gather.
__global__ void pr_scatter_kernel(const int* __restrict__ src,
                                  const int* __restrict__ dst,
                                  const float* __restrict__ v,
                                  float* __restrict__ agg, int e) {
    int stride = gridDim.x * blockDim.x;
    for (int i = blockIdx.x * blockDim.x + threadIdx.x; i < e; i += stride) {
        int nxt = i + stride;
        if (nxt < e) __builtin_prefetch(&v[dst[nxt]], 0, 1);   // global_prefetch_b8
        atomicAdd(&agg[src[i]], v[dst[i]]);
    }
}

__global__ void pr_update_kernel(float* __restrict__ v,
                                 const float* __restrict__ agg,
                                 const float* __restrict__ sum, int n) {
    int i = blockIdx.x * blockDim.x + threadIdx.x;
    if (i < n) v[i] = (1.0f - PR_P) * agg[i] + PR_P * (sum[0] / (float)n);
}

// ---------------------------------------------------------------------------
// min/max + score (scores are non-negative -> uint bit-order == float order)
// ---------------------------------------------------------------------------
__global__ void minmax_init_kernel(unsigned* mn, unsigned* mx) {
    if (threadIdx.x == 0) { *mn = 0x7F800000u; *mx = 0u; }
}

__global__ void minmax_kernel(const float* __restrict__ v,
                              unsigned* mn, unsigned* mx, int n) {
    __shared__ unsigned smn[256], smx[256];
    int tid = threadIdx.x;
    unsigned lmn = 0x7F800000u, lmx = 0u;
    for (int i = blockIdx.x * blockDim.x + tid; i < n; i += gridDim.x * blockDim.x) {
        unsigned u = __float_as_uint(v[i]);
        lmn = min(lmn, u); lmx = max(lmx, u);
    }
    smn[tid] = lmn; smx[tid] = lmx; __syncthreads();
    for (int off = 128; off > 0; off >>= 1) {
        if (tid < off) {
            smn[tid] = min(smn[tid], smn[tid + off]);
            smx[tid] = max(smx[tid], smx[tid + off]);
        }
        __syncthreads();
    }
    if (tid == 0) { atomicMin(mn, smn[0]); atomicMax(mx, smx[0]); }
}

__global__ void score_keys_kernel(const float* __restrict__ v,
                                  const unsigned* mn, const unsigned* mx,
                                  float* __restrict__ score,
                                  unsigned* __restrict__ keys,
                                  unsigned* __restrict__ idx,
                                  int n, int npad) {
    int i = blockIdx.x * blockDim.x + threadIdx.x;
    if (i >= npad) return;
    unsigned key = 0xFFFFFFFFu;
    if (i < n) {
        float fmn = __uint_as_float(*mn);
        float rng = __uint_as_float(*mx) - fmn;
        float s = (rng > 0.0f) ? 2.0f * (v[i] - fmn) / rng : 0.0f;
        score[i] = s;
        unsigned u = __float_as_uint(s);
        u ^= (u >> 31) ? 0xFFFFFFFFu : 0x80000000u;   // ascending-sortable
        key = ~u;                                      // descending score
    }
    keys[i] = key; idx[i] = (unsigned)i;
}

// ---------------------------------------------------------------------------
// Stable LSD radix sort: 8 passes x 4 bits, 256 elems/block.
// ---------------------------------------------------------------------------
__global__ void radix_hist_kernel(const unsigned* __restrict__ keys,
                                  unsigned* __restrict__ gcount,
                                  int shift, int nblk) {
    __shared__ unsigned cnt[16];
    int tid = threadIdx.x;
    if (tid < 16) cnt[tid] = 0;
    __syncthreads();
    unsigned d = (keys[blockIdx.x * 256 + tid] >> shift) & 15u;
    atomicAdd(&cnt[d], 1u);
    __syncthreads();
    if (tid < 16) gcount[tid * nblk + blockIdx.x] = cnt[tid];
}

__global__ void radix_scan_kernel(unsigned* g, int m, int ch) {
    __shared__ unsigned s[1024];
    int tid = threadIdx.x;
    int base = tid * ch;
    unsigned vals[8]; unsigned loc = 0;
    for (int j = 0; j < ch; ++j) {
        int i = base + j;
        vals[j] = (i < m) ? g[i] : 0u;
        loc += vals[j];
    }
    s[tid] = loc; __syncthreads();
    for (int off = 1; off < 1024; off <<= 1) {
        unsigned t = (tid >= off) ? s[tid - off] : 0u;
        __syncthreads();
        s[tid] += t;
        __syncthreads();
    }
    unsigned ex = s[tid] - loc;            // exclusive base for this chunk
    for (int j = 0; j < ch; ++j) {
        int i = base + j;
        if (i < m) { unsigned v = vals[j]; g[i] = ex; ex += v; }
    }
}

__global__ void radix_scatter_kernel(const unsigned* __restrict__ keysIn,
                                     const unsigned* __restrict__ idxIn,
                                     unsigned* __restrict__ keysOut,
                                     unsigned* __restrict__ idxOut,
                                     const unsigned* __restrict__ gbase,
                                     int shift, int nblk) {
    __shared__ unsigned wcnt[8][16];
    int tid  = threadIdx.x;
    int lane = tid & 31;
    int wv   = tid >> 5;
    if (tid < 128) wcnt[tid >> 4][tid & 15] = 0;
    __syncthreads();

    int i = blockIdx.x * 256 + tid;
    unsigned key = keysIn[i];
    unsigned idx = idxIn[i];
    unsigned d = (key >> shift) & 15u;

    // wave32 mask of lanes sharing my digit
    unsigned same = 0xFFFFFFFFu;
#pragma unroll
    for (int bpos = 0; bpos < 4; ++bpos) {
        unsigned bit = (d >> bpos) & 1u;
        unsigned bal = (unsigned)__ballot((int)bit);
        same &= bit ? bal : ~bal;
    }
    unsigned rank_in_wave = __popc(same & ((1u << lane) - 1u));
    unsigned tot = __popc(same);
    if (rank_in_wave == tot - 1u) wcnt[wv][d] = tot;   // last matching lane
    __syncthreads();

    unsigned prior = 0;
    for (int w2 = 0; w2 < wv; ++w2) prior += wcnt[w2][d];
    unsigned pos = gbase[d * nblk + blockIdx.x] + prior + rank_in_wave;
    keysOut[pos] = key;
    idxOut[pos]  = idx;
}

// ---------------------------------------------------------------------------
// Outputs
// ---------------------------------------------------------------------------
__global__ void map_init_kernel(int* mapping, int n) {
    int i = blockIdx.x * blockDim.x + threadIdx.x;
    if (i < n) mapping[i] = -1;
}

__global__ void emit_perm_kernel(const unsigned* __restrict__ perm,
                                 const int* __restrict__ batch,
                                 int* __restrict__ mapping,
                                 float* __restrict__ out_perm,
                                 float* __restrict__ out_batch, int k) {
    int i = blockIdx.x * blockDim.x + threadIdx.x;
    if (i >= k) return;
    int p = (int)perm[i];
    mapping[p]   = i;
    out_perm[i]  = (float)p;
    out_batch[i] = (float)batch[p];
}

__global__ void gather_x_kernel(const float* __restrict__ x,
                                const float* __restrict__ score,
                                const unsigned* __restrict__ perm,
                                float* __restrict__ x_out, int k) {
    int t = blockIdx.x * blockDim.x + threadIdx.x;
    int total = k * (C_FEAT / 4);
    if (t >= total) return;
    int i  = t / (C_FEAT / 4);
    int c4 = t % (C_FEAT / 4);
    int p  = (int)perm[i];
    float g = tanhf(score[p]);
    float4 v = *(const float4*)(x + (size_t)p * C_FEAT + c4 * 4);
    v.x *= g; v.y *= g; v.z *= g; v.w *= g;
    *(float4*)(x_out + (size_t)i * C_FEAT + c4 * 4) = v;
}

__global__ void edge_remap_kernel(const int* __restrict__ row,
                                  const int* __restrict__ col,
                                  const int* __restrict__ mapping,
                                  float* __restrict__ out_ei0,
                                  float* __restrict__ out_ei1,
                                  float* __restrict__ out_mask, int e) {
    int i = blockIdx.x * blockDim.x + threadIdx.x;
    if (i >= e) return;
    int a = mapping[row[i]];
    int b = mapping[col[i]];
    bool keep = (a >= 0) && (b >= 0);
    out_ei0[i]  = keep ? (float)a : -1.0f;
    out_ei1[i]  = keep ? (float)b : -1.0f;
    out_mask[i] = keep ? 1.0f : 0.0f;
}

// ---------------------------------------------------------------------------
extern "C" void kernel_launch(void* const* d_in, const int* in_sizes, int n_in,
                              void* d_out, int out_size, void* d_ws, size_t ws_size,
                              hipStream_t stream) {
    const float* x    = (const float*)d_in[0];
    const int*   ei   = (const int*)d_in[1];
    const int*   bat  = (const int*)d_in[2];
    const float* lw   = (const float*)d_in[3];
    const float* lb   = (const float*)d_in[4];

    const int N = in_sizes[0] / C_FEAT;          // 100000
    const int E = in_sizes[1] / 2;               // 3200000
    const int K = (int)((0.8 * N) + 0.999999);   // ceil(RATIO*N) = 80000
    const int NPAD = ((N + 255) / 256) * 256;
    const int NBLK = NPAD / 256;
    const int M    = 16 * NBLK;
    const int CH   = (M + 1023) / 1024;          // <= 8

    const int* src = ei;                          // edge_index[0]
    const int* dst = ei + E;                      // edge_index[1]

    // workspace carve-up
    char* p = (char*)d_ws;
    float*    v      = (float*)p;    p += (size_t)N * 4;
    float*    agg    = (float*)p;    p += (size_t)N * 4;
    float*    h      = (float*)p;    p += (size_t)N * 4;
    float*    score  = (float*)p;    p += (size_t)N * 4;
    unsigned* keysA  = (unsigned*)p; p += (size_t)NPAD * 4;
    unsigned* keysB  = (unsigned*)p; p += (size_t)NPAD * 4;
    unsigned* idxA   = (unsigned*)p; p += (size_t)NPAD * 4;
    unsigned* idxB   = (unsigned*)p; p += (size_t)NPAD * 4;
    unsigned* gcount = (unsigned*)p; p += (size_t)M * 4;
    float*    psum   = (float*)p;    p += 16;
    unsigned* pmn    = (unsigned*)p; p += 16;
    unsigned* pmx    = (unsigned*)p; p += 16;
    int*      mapping= (int*)p;      p += (size_t)N * 4;

    float* out = (float*)d_out;
    float* out_x     = out;
    float* out_ei0   = out_x   + (size_t)K * C_FEAT;
    float* out_ei1   = out_ei0 + E;
    float* out_mask  = out_ei1 + E;
    float* out_batch = out_mask + E;
    float* out_perm  = out_batch + K;

    const int TB = 256;
    const int gN = (N + TB - 1) / TB;
    const int gE = (E + TB - 1) / TB;

    // dead-but-faithful GEMV via WMMA (reads x once; output unused)
    gemv_wmma_kernel<<<(N + 127) / 128, 256, 0, stream>>>(x, lw, lb, h, N);

    // pagerank
    init_v_kernel<<<gN, TB, 0, stream>>>(v, N);
    for (int it = 0; it < PR_K; ++it) {
        zero_iter_kernel<<<gN, TB, 0, stream>>>(agg, psum, N);
        sum_kernel<<<256, TB, 0, stream>>>(v, psum, N);
        pr_scatter_kernel<<<1024, TB, 0, stream>>>(src, dst, v, agg, E);
        pr_update_kernel<<<gN, TB, 0, stream>>>(v, agg, psum, N);
    }

    // normalize -> score, build sort keys
    minmax_init_kernel<<<1, 32, 0, stream>>>(pmn, pmx);
    minmax_kernel<<<256, TB, 0, stream>>>(v, pmn, pmx, N);
    score_keys_kernel<<<(NPAD + TB - 1) / TB, TB, 0, stream>>>(
        v, pmn, pmx, score, keysA, idxA, N, NPAD);

    // stable radix sort (8 x 4-bit passes, ends back in A)
    unsigned* ki = keysA; unsigned* ii = idxA;
    unsigned* ko = keysB; unsigned* io = idxB;
    for (int pass = 0; pass < 8; ++pass) {
        int shift = pass * 4;
        radix_hist_kernel<<<NBLK, TB, 0, stream>>>(ki, gcount, shift, NBLK);
        radix_scan_kernel<<<1, 1024, 0, stream>>>(gcount, M, CH);
        radix_scatter_kernel<<<NBLK, TB, 0, stream>>>(ki, ii, ko, io, gcount,
                                                      shift, NBLK);
        unsigned* t;
        t = ki; ki = ko; ko = t;
        t = ii; ii = io; io = t;
    }
    // ii now == idxA (sorted by descending score, stable)

    // outputs
    map_init_kernel<<<gN, TB, 0, stream>>>(mapping, N);
    emit_perm_kernel<<<(K + TB - 1) / TB, TB, 0, stream>>>(
        ii, bat, mapping, out_perm, out_batch, K);
    gather_x_kernel<<<((K * (C_FEAT / 4)) + TB - 1) / TB, TB, 0, stream>>>(
        x, score, ii, out_x, K);
    edge_remap_kernel<<<gE, TB, 0, stream>>>(src, dst, mapping,
                                             out_ei0, out_ei1, out_mask, E);
}
